// SelfAttention_798863917115
// MI455X (gfx1250) — matrix operand order repaired and checked by reading, not compiled, not run on hardware
//
#include <hip/hip_runtime.h>
#include <hip/hip_bf16.h>
#include <math.h>

#define NB 8
#define NS 2048
#define ND 512

typedef __attribute__((ext_vector_type(16))) __bf16 bf16x16;
typedef __attribute__((ext_vector_type(8)))  __bf16 bf16x8;
typedef __attribute__((ext_vector_type(8)))  float  floatx8;
typedef int v4i_vs __attribute__((vector_size(16)));   // matches async-LDS builtin param type

#if __has_builtin(__builtin_amdgcn_global_load_async_to_lds_b128) && \
    __has_builtin(__builtin_amdgcn_s_wait_asynccnt)
#define USE_ASYNC_LDS 1
#else
#define USE_ASYNC_LDS 0
#endif

__device__ __forceinline__ __bf16 f2bf(float f) {
  unsigned u = __builtin_bit_cast(unsigned, f);
  u += 0x7fffu + ((u >> 16) & 1u);                 // round-to-nearest-even
  unsigned short h = (unsigned short)(u >> 16);
  return __builtin_bit_cast(__bf16, h);
}

// A fragment: 16x32 bf16, row m = lane&15; per-lane K chunks at half*8 and 16+half*8
__device__ __forceinline__ bf16x16 frag_a(const __bf16* base, int stride, int lane) {
  const int r = lane & 15, h = lane >> 4;
  const __bf16* p = base + (size_t)r * stride + h * 8;
  bf16x8 lo = *(const bf16x8*)(p);
  bf16x8 hi = *(const bf16x8*)(p + 16);
  bf16x16 o;
#pragma unroll
  for (int i = 0; i < 8; ++i) { o[i] = lo[i]; o[i + 8] = hi[i]; }
  return o;
}

// B fragment: 32x16 bf16, col n = lane&15; per-lane K chunks at half*8 and 16+half*8
__device__ __forceinline__ bf16x16 frag_b(const __bf16* base, int stride, int lane) {
  const int n = lane & 15, h = lane >> 4;
  const __bf16* p = base + (size_t)n * stride + h * 8;
  bf16x8 lo = *(const bf16x8*)(p);
  bf16x8 hi = *(const bf16x8*)(p + 16);
  bf16x16 o;
#pragma unroll
  for (int i = 0; i < 8; ++i) { o[i] = lo[i]; o[i + 8] = hi[i]; }
  return o;
}

__device__ __forceinline__ floatx8 wmma_bf16(bf16x16 a, bf16x16 b, floatx8 c) {
  return __builtin_amdgcn_wmma_f32_16x16x32_bf16(false, a, false, b, (short)0, c, false, false);
}

// ---------------- Kernel 1: fp32 x -> bf16 ----------------
__global__ __launch_bounds__(256) void sa_cvt_x(const float* __restrict__ x,
                                                __bf16* __restrict__ xb) {
  size_t i = ((size_t)blockIdx.x * 256 + threadIdx.x) * 8;
  float4 f0 = *(const float4*)(x + i);
  float4 f1 = *(const float4*)(x + i + 4);
  bf16x8 o;
  o[0] = f2bf(f0.x); o[1] = f2bf(f0.y); o[2] = f2bf(f0.z); o[3] = f2bf(f0.w);
  o[4] = f2bf(f1.x); o[5] = f2bf(f1.y); o[6] = f2bf(f1.z); o[7] = f2bf(f1.w);
  *(bf16x8*)(xb + i) = o;
}

// ---------------- Kernel 2: W[k][n] -> Wt[n][k] bf16 (LDS tile transpose) ----------------
__global__ __launch_bounds__(256) void sa_transpose_w(
    const float* __restrict__ wq, const float* __restrict__ wk, const float* __restrict__ wv,
    __bf16* __restrict__ tq, __bf16* __restrict__ tk, __bf16* __restrict__ tv) {
  __shared__ float tile[32][33];
  const float* src = (blockIdx.z == 0) ? wq : (blockIdx.z == 1) ? wk : wv;
  __bf16* dst      = (blockIdx.z == 0) ? tq : (blockIdx.z == 1) ? tk : tv;
  const int k0 = blockIdx.x * 32, n0 = blockIdx.y * 32;
  const int tx = threadIdx.x, ty = threadIdx.y;
#pragma unroll
  for (int r = ty; r < 32; r += 8) tile[r][tx] = src[(size_t)(k0 + r) * ND + n0 + tx];
  __syncthreads();
#pragma unroll
  for (int r = ty; r < 32; r += 8) dst[(size_t)(n0 + r) * ND + k0 + tx] = f2bf(tile[tx][r]);
}

// ---------------- Kernel 3: projections Q,K (row-major) and V^T [b][d][s] ----------------
__global__ __launch_bounds__(256) void sa_proj(
    const __bf16* __restrict__ xb,
    const __bf16* __restrict__ wqt, const __bf16* __restrict__ wkt, const __bf16* __restrict__ wvt,
    const float* __restrict__ bq, const float* __restrict__ bk, const float* __restrict__ bv,
    __bf16* __restrict__ Qb, __bf16* __restrict__ Kb, __bf16* __restrict__ Vtb) {
  const int tid  = threadIdx.x;
  const int lane = tid & 31;
  const int wave = tid >> 5;            // 8 waves; wave == N-group
  const int which = blockIdx.x >> 10;   // 0=Q 1=K 2=V
  const int mt    = blockIdx.x & 1023;  // M tile (16 rows of B*S)
  const int mA = mt * 16;
  const int n0 = wave * 64;

  const __bf16* wt   = (which == 0) ? wqt : (which == 1) ? wkt : wvt;
  const float*  bias = (which == 0) ? bq  : (which == 1) ? bk  : bv;

  floatx8 c[4];
#pragma unroll
  for (int t = 0; t < 4; ++t) c[t] = (floatx8)0.0f;

  const __bf16* abase = xb + (size_t)mA * ND;

  // software-pipelined K loop: prefetch iter kc+1 fragments before consuming iter kc
  bf16x16 a = frag_a(abase, ND, lane);
  bf16x16 bfr[4];
#pragma unroll
  for (int t = 0; t < 4; ++t) bfr[t] = frag_b(wt + (size_t)(n0 + t * 16) * ND, ND, lane);

#pragma unroll
  for (int kc = 0; kc < 16; ++kc) {
    bf16x16 a_n = a;
    bf16x16 b_n[4];
#pragma unroll
    for (int t = 0; t < 4; ++t) b_n[t] = bfr[t];
    if (kc < 15) {
      a_n = frag_a(abase + (kc + 1) * 32, ND, lane);
#pragma unroll
      for (int t = 0; t < 4; ++t)
        b_n[t] = frag_b(wt + (size_t)(n0 + t * 16) * ND + (kc + 1) * 32, ND, lane);
    }
#pragma unroll
    for (int t = 0; t < 4; ++t) c[t] = wmma_bf16(a, bfr[t], c[t]);
    a = a_n;
#pragma unroll
    for (int t = 0; t < 4; ++t) bfr[t] = b_n[t];
  }

  const int l16 = lane & 15, half = lane >> 4;
#pragma unroll
  for (int t = 0; t < 4; ++t) {
    const int n = n0 + t * 16 + l16;
    const float bi = bias[n];
#pragma unroll
    for (int v = 0; v < 8; ++v) {
      const float val = c[t][v] + bi;
      const int row = mA + v + 8 * half;            // global row over B*S
      if (which < 2) {
        __bf16* dst = which ? Kb : Qb;
        dst[(size_t)row * ND + n] = f2bf(val);
      } else {
        const int bidx = row >> 11;                 // / NS
        const int s    = row & (NS - 1);
        Vtb[((size_t)bidx * ND + n) * NS + s] = f2bf(val);
      }
    }
  }
}

// ---------------- Kernel 4: flash attention ----------------
// block = 256 (8 waves) handles (batch b, 16 query rows); wave w owns O cols [w*64, w*64+64)
__global__ __launch_bounds__(256, 2) void sa_attn(
    const __bf16* __restrict__ Qb, const __bf16* __restrict__ Kb,
    const __bf16* __restrict__ Vtb, float* __restrict__ out) {
  __shared__ __align__(16) __bf16 Qs[16 * ND];      // 16 KB
  __shared__ __align__(16) __bf16 Ps[16 * 128];     // 4 KB
  __shared__ float mw[8 * 16];
  __shared__ float lw[8 * 16];
  __shared__ float mrow[16], lrow[16], arow[16];

  const int tid  = threadIdx.x;
  const int lane = tid & 31;
  const int wave = tid >> 5;
  const int half = lane >> 4;
  const int l16  = lane & 15;

  const int b  = blockIdx.x >> 7;
  const int q0 = (blockIdx.x & 127) << 4;

  // stage Q tile into LDS (16x512 bf16 = 16 KB)
#if USE_ASYNC_LDS
  {
    const char* g = (const char*)(Qb + (size_t)(b * NS + q0) * ND);
    char* l = (char*)Qs;
#pragma unroll
    for (int i = tid; i < 1024; i += 256) {
      __builtin_amdgcn_global_load_async_to_lds_b128(
          (v4i_vs*)(g + i * 16), (v4i_vs*)(l + i * 16), 0, 0);
    }
    __builtin_amdgcn_s_wait_asynccnt(0);
  }
#else
  {
    const uint4* src = (const uint4*)(Qb + (size_t)(b * NS + q0) * ND);
    uint4* dst = (uint4*)Qs;
#pragma unroll
    for (int i = tid; i < 1024; i += 256) dst[i] = src[i];
  }
#endif
  if (tid < 16) { mrow[tid] = -3.0e38f; lrow[tid] = 0.0f; }
  __syncthreads();

  floatx8 o[4];
#pragma unroll
  for (int t = 0; t < 4; ++t) o[t] = (floatx8)0.0f;

  const float scale = 0.044194173824159216f;        // 1/sqrt(512)

  for (int kb = 0; kb < NS / 128; ++kb) {
    const int keyBase = kb * 128;

    // prefetch next key block's K rows and V columns for this wave
    if (kb + 1 < NS / 128) {
      const char* nk = (const char*)(Kb + (size_t)(b * NS + keyBase + 128 + wave * 16) * ND);
#pragma unroll
      for (int i = 0; i < 4; ++i) __builtin_prefetch(nk + lane * 128 + i * 4096, 0, 3);
      const char* nv = (const char*)(Vtb + (size_t)b * ND * NS + (size_t)(wave * 64) * NS +
                                     keyBase + 128);
#pragma unroll
      for (int i = 0; i < 2; ++i) __builtin_prefetch(nv + (size_t)(lane + i * 32) * (NS * 2), 0, 3);
    }

    // ---- scores: this wave's 16 keys, full D reduction (pipelined) ----
    floatx8 c = (floatx8)0.0f;
    const __bf16* Kgw = Kb + (size_t)(b * NS + keyBase + wave * 16) * ND;
    bf16x16 qa = frag_a(Qs, ND, lane);              // LDS
    bf16x16 kf = frag_b(Kgw, ND, lane);             // global
#pragma unroll
    for (int kc = 0; kc < 16; ++kc) {
      bf16x16 qn = qa, kn = kf;
      if (kc < 15) {
        qn = frag_a(Qs + (kc + 1) * 32, ND, lane);
        kn = frag_b(Kgw + (kc + 1) * 32, ND, lane);
      }
      c = wmma_bf16(qa, kf, c);
      qa = qn; kf = kn;
    }
#pragma unroll
    for (int v = 0; v < 8; ++v) c[v] *= scale;

    // row-max across the 16 lanes holding each row
    floatx8 r = c;
#pragma unroll
    for (int m = 1; m <= 8; m <<= 1) {
#pragma unroll
      for (int v = 0; v < 8; ++v) r[v] = fmaxf(r[v], __shfl_xor(r[v], m, 32));
    }
    if (l16 == 0) {
#pragma unroll
      for (int v = 0; v < 8; ++v) mw[wave * 16 + v + 8 * half] = r[v];
    }
    __syncthreads();

    if (tid < 16) {
      float mo = mrow[tid], mn = mo;
#pragma unroll
      for (int w = 0; w < 8; ++w) mn = fmaxf(mn, mw[w * 16 + tid]);
      arow[tid] = __expf(mo - mn);
      mrow[tid] = mn;
    }
    __syncthreads();

    // ---- P = exp(s - m_new); write bf16 P to LDS; row-sums ----
    floatx8 pv;
#pragma unroll
    for (int v = 0; v < 8; ++v) {
      const int m = v + 8 * half;
      const float p = __expf(c[v] - mrow[m]);
      pv[v] = p;
      Ps[m * 128 + wave * 16 + l16] = f2bf(p);
    }
    floatx8 sv = pv;
#pragma unroll
    for (int m = 1; m <= 8; m <<= 1) {
#pragma unroll
      for (int v = 0; v < 8; ++v) sv[v] += __shfl_xor(sv[v], m, 32);
    }
    if (l16 == 0) {
#pragma unroll
      for (int v = 0; v < 8; ++v) lw[wave * 16 + v + 8 * half] = sv[v];
    }
    __syncthreads();

    if (tid < 16) {
      float acc = lrow[tid] * arow[tid];
#pragma unroll
      for (int w = 0; w < 8; ++w) acc += lw[w * 16 + tid];
      lrow[tid] = acc;
    }

    // ---- rescale O, accumulate P·V for this wave's 64 columns (pipelined) ----
#pragma unroll
    for (int v = 0; v < 8; ++v) {
      const float al = arow[v + 8 * half];
#pragma unroll
      for (int t = 0; t < 4; ++t) o[t][v] *= al;
    }
    const __bf16* Vg = Vtb + (size_t)b * ND * NS + (size_t)(wave * 64) * NS + keyBase;
    bf16x16 pa = frag_a(Ps, 128, lane);             // LDS P
    bf16x16 vf = frag_b(Vg, NS, lane);              // global V^T
#pragma unroll
    for (int step = 0; step < 16; ++step) {
      const int kc = step >> 2, t = step & 3;
      bf16x16 pn = pa, vn = vf;
      if (step < 15) {
        const int ns2 = step + 1, nkc = ns2 >> 2, nt = ns2 & 3;
        vn = frag_b(Vg + (size_t)(nt * 16) * NS + nkc * 32, NS, lane);
        if (nt == 0) pn = frag_a(Ps + nkc * 32, 128, lane);
      }
      o[t] = wmma_bf16(pa, vf, o[t]);
      pa = pn; vf = vn;
      (void)kc;
    }
    __syncthreads();
  }

  // ---- epilogue: normalize and store fp32 ----
#pragma unroll
  for (int t = 0; t < 4; ++t) {
#pragma unroll
    for (int v = 0; v < 8; ++v) {
      const int m = v + 8 * half;
      const float val = o[t][v] / lrow[m];
      const int dcol = wave * 64 + t * 16 + l16;
      out[((size_t)(b * NS + q0 + m)) * ND + dcol] = val;
    }
  }
}

// ---------------- launch ----------------
extern "C" void kernel_launch(void* const* d_in, const int* in_sizes, int n_in,
                              void* d_out, int out_size, void* d_ws, size_t ws_size,
                              hipStream_t stream) {
  (void)in_sizes; (void)n_in; (void)out_size; (void)ws_size;
  const float* x  = (const float*)d_in[0];
  const float* Wq = (const float*)d_in[1];
  const float* bq = (const float*)d_in[2];
  const float* Wk = (const float*)d_in[3];
  const float* bk = (const float*)d_in[4];
  const float* Wv = (const float*)d_in[5];
  const float* bv = (const float*)d_in[6];

  char* ws = (char*)d_ws;
  const size_t SZ = (size_t)NB * NS * ND * sizeof(__bf16);   // 16 MB
  __bf16* xb  = (__bf16*)(ws);
  __bf16* Qb  = (__bf16*)(ws + SZ);
  __bf16* Kb  = (__bf16*)(ws + 2 * SZ);
  __bf16* Vtb = (__bf16*)(ws + 3 * SZ);
  __bf16* wqt = (__bf16*)(ws + 4 * SZ);
  __bf16* wkt = (__bf16*)(ws + 4 * SZ + (size_t)ND * ND * 2);
  __bf16* wvt = (__bf16*)(ws + 4 * SZ + (size_t)ND * ND * 4);

  // 1. x -> bf16
  sa_cvt_x<<<(NB * NS * ND) / (256 * 8), 256, 0, stream>>>(x, xb);
  // 2. weight transpose+convert
  dim3 tg(ND / 32, ND / 32, 3), tb(32, 8);
  sa_transpose_w<<<tg, tb, 0, stream>>>(Wq, Wk, Wv, wqt, wkt, wvt);
  // 3. projections (3 outputs * 1024 M-tiles; 8 waves = 8 N-groups per block)
  sa_proj<<<3 * 1024, 256, 0, stream>>>(xb, wqt, wkt, wvt, bq, bk, bv, Qb, Kb, Vtb);
  // 4. flash attention
  sa_attn<<<NB * (NS / 16), 256, 0, stream>>>(Qb, Kb, Vtb, (float*)d_out);
}
